// Attention_Critic_66975720013857
// MI455X (gfx1250) — matrix operand order repaired and checked by reading, not compiled, not run on hardware
//
#include <hip/hip_runtime.h>

// ---------------------------------------------------------------------------
// Problem constants (from the reference)
// ---------------------------------------------------------------------------
#define NA 3
#define BATCH 32768
#define H 128
#define SDIM 18
#define ADIM 2
#define NCOLS 20            // 18 s-columns + 2 a-columns per agent
#define LRELU_SLOPE 0.01f
#define BN_EPS 1e-5f
#define ATTN_SCALE 0.08838834764831845f   // 1/sqrt(128)

typedef __bf16 bf16;
typedef __attribute__((ext_vector_type(16))) __bf16 v16bf;
typedef __attribute__((ext_vector_type(8)))  __bf16 v8bf;
typedef __attribute__((ext_vector_type(8)))  float   v8f;

__device__ __forceinline__ float lrelu(float x) {
    return x > 0.0f ? x : LRELU_SLOPE * x;
}

// ---------------------------------------------------------------------------
// 1) BatchNorm stats: mean / rstd per (agent, column) over the batch.
//    col 0..17 -> s[:, :, col], col 18..19 -> a[:, :, col-18]
// ---------------------------------------------------------------------------
__global__ __launch_bounds__(256)
void bn_stats_kernel(const float* __restrict__ s, const float* __restrict__ a,
                     float* __restrict__ mean, float* __restrict__ rstd) {
    int col = blockIdx.x % NCOLS;
    int n   = blockIdx.x / NCOLS;
    const float* src;
    long stride;
    if (col < SDIM) { src = s + (long)n * BATCH * SDIM + col; stride = SDIM; }
    else            { src = a + (long)n * BATCH * ADIM + (col - SDIM); stride = ADIM; }

    float sum = 0.f, sq = 0.f;
    for (int b = threadIdx.x; b < BATCH; b += 256) {
        float v = src[(long)b * stride];
        sum += v; sq += v * v;
    }
    __shared__ float s1[256], s2[256];
    s1[threadIdx.x] = sum; s2[threadIdx.x] = sq;
    __syncthreads();
    for (int w = 128; w > 0; w >>= 1) {
        if (threadIdx.x < w) { s1[threadIdx.x] += s1[threadIdx.x + w];
                               s2[threadIdx.x] += s2[threadIdx.x + w]; }
        __syncthreads();
    }
    if (threadIdx.x == 0) {
        float m = s1[0] / (float)BATCH;
        float v = s2[0] / (float)BATCH - m * m;   // biased var (jnp.var default)
        mean[n * NCOLS + col] = m;
        rstd[n * NCOLS + col] = rsqrtf(v + BN_EPS);
    }
}

// ---------------------------------------------------------------------------
// 2) Weight convert + transpose: f32 [K x N] row-major -> bf16 [N x K]
//    (column-major of original) so WMMA B-fragments are contiguous loads.
// ---------------------------------------------------------------------------
__global__ __launch_bounds__(256)
void wtrans_kernel(const float* __restrict__ W, bf16* __restrict__ Wt,
                   int K, int N, long w_stride, long wt_stride) {
    int mat = blockIdx.z;
    const float* src = W + (long)mat * w_stride;
    bf16* dst = Wt + (long)mat * wt_stride;
    int idx = blockIdx.x * 256 + threadIdx.x;
    if (idx < K * N) {
        int k = idx / N, n = idx % N;
        dst[(long)n * K + k] = (bf16)src[idx];
    }
}

// ---------------------------------------------------------------------------
// 3) Small-K encoders (en / oa / goal / s_enc) with fused BN + bias + lrelu.
//    One block = one (agent, batch) row, 128 threads over H.
//    en_enc -> merge_in[:, 0:H]; s_enc -> cw1_in[:, 0:H].
// ---------------------------------------------------------------------------
__global__ __launch_bounds__(128)
void encode_kernel(const float* __restrict__ s, const float* __restrict__ a,
                   const float* __restrict__ mean, const float* __restrict__ rstd,
                   const float* __restrict__ en_W, const float* __restrict__ en_b,
                   const float* __restrict__ oa_W, const float* __restrict__ oa_b,
                   const float* __restrict__ goal_W, const float* __restrict__ goal_b,
                   const float* __restrict__ senc_W, const float* __restrict__ senc_b,
                   bf16* __restrict__ merge_in,   // [NA*B][3H]
                   bf16* __restrict__ oa_enc,     // [NA*2][B][H]
                   bf16* __restrict__ g_enc,      // [NA*3][B][H]
                   bf16* __restrict__ cw1_in) {   // [NA*B][2H]
    long nb = blockIdx.x;
    int n = (int)(nb / BATCH);
    long b = nb % BATCH;
    int h = threadIdx.x;

    const float* srow = s + nb * SDIM;
    const float* arow = a + nb * ADIM;
    const float* mn = mean + n * NCOLS;
    const float* rs = rstd + n * NCOLS;

    float shat[SDIM];
#pragma unroll
    for (int i = 0; i < SDIM; i++) shat[i] = (srow[i] - mn[i]) * rs[i];
    float ahat[ADIM];
#pragma unroll
    for (int i = 0; i < ADIM; i++) ahat[i] = (arow[i] - mn[SDIM + i]) * rs[SDIM + i];

    // entity encoder: [s0..3, a0..1] @ en_W[n]  (6 x H)
    float acc = en_b[n * H + h];
#pragma unroll
    for (int i = 0; i < 4; i++) acc += shat[i] * en_W[((long)n * 6 + i) * H + h];
#pragma unroll
    for (int i = 0; i < 2; i++) acc += ahat[i] * en_W[((long)n * 6 + 4 + i) * H + h];
    merge_in[nb * (3 * H) + h] = (bf16)lrelu(acc);

    // other-agent encoders (j = 0,1): s cols 4+4j..7+4j @ oa_W[n] (4 x H)
#pragma unroll
    for (int j = 0; j < 2; j++) {
        float ac = oa_b[n * H + h];
#pragma unroll
        for (int i = 0; i < 4; i++)
            ac += shat[4 + 4 * j + i] * oa_W[((long)n * 4 + i) * H + h];
        oa_enc[(((long)(n * 2 + j)) * BATCH + b) * H + h] = (bf16)lrelu(ac);
    }

    // goal encoders (j = 0..2): s cols 12+2j..13+2j @ goal_W[n] (2 x H)
#pragma unroll
    for (int j = 0; j < 3; j++) {
        float ac = goal_b[n * H + h];
#pragma unroll
        for (int i = 0; i < 2; i++)
            ac += shat[12 + 2 * j + i] * goal_W[((long)n * 2 + i) * H + h];
        g_enc[(((long)(n * 3 + j)) * BATCH + b) * H + h] = (bf16)lrelu(ac);
    }

    // critic state encoder: full s (18 x H)
    float sc = senc_b[n * H + h];
#pragma unroll
    for (int i = 0; i < SDIM; i++) sc += shat[i] * senc_W[((long)n * SDIM + i) * H + h];
    cw1_in[nb * (2 * H) + h] = (bf16)lrelu(sc);
}

// ---------------------------------------------------------------------------
// 4) WMMA bf16 GEMM:  C[rows x 128] = act(A[rows x K] * Wt^T + bias)
//    Wt is [128 x K] (column-major of original W).
//
//    Block = 128 threads = 4 waves; block owns 128 rows, each wave 32 rows
//    (two 16-row accumulator banks -> 2x arithmetic intensity per B read).
//    The 128x128 bf16 weight panel is staged cooperatively in LDS (32 KB,
//    row stride padded to 136 bf16 so the 16 lanes of a B-fragment read hit
//    distinct banks).  B fragments then come from ds_load (short latency,
//    DScnt) instead of per-WMMA global-load drains; A loads stay on the
//    vmem path and overlap with the WMMA burst.
//
//    A fragment (16x32 bf16, per-lane, contiguous K order used here):
//      lanes 0-15  : row M=lane,    K = kb..kb+15
//      lanes 16-31 : row M=lane-16, K = kb+16..kb+31
//    B fragment (32x16): lane = column, 16 contiguous K values
//      (lanes 0-15: kb..kb+15, lanes 16-31: kb+16..kb+31)
// ---------------------------------------------------------------------------
#define LDSW_STRIDE 136     // 128 + 8 bf16 pad (16B) -> bank-conflict-free

template <int K>
__global__ __launch_bounds__(128)
void gemm_bf16_kernel(const bf16* __restrict__ A, long lda,
                      const bf16* __restrict__ Wt, const float* __restrict__ bias,
                      long rows_per_group, long wt_group_stride, long bias_group_stride,
                      bf16* __restrict__ C, long ldc, int do_lrelu) {
    __shared__ bf16 ldsw[128 * LDSW_STRIDE];

    int tid  = threadIdx.x;
    int lane = tid & 31;
    int wave = tid >> 5;
    long block_m = (long)blockIdx.x * 128;
    long tile_m  = block_m + (long)wave * 32;
    long grp = block_m / rows_per_group;   // rows_per_group is a multiple of 128
    const bf16* wt = Wt + grp * wt_group_stride;
    const float* bs = bias ? (bias + grp * bias_group_stride) : nullptr;

    int hl  = lane >> 4;     // which 16-lane half
    int l15 = lane & 15;
    const bf16* arow0 = A + (tile_m + l15) * lda;
    const bf16* arow1 = A + (tile_m + 16 + l15) * lda;

    v8f acc0[8] = {};
    v8f acc1[8] = {};

#pragma unroll
    for (int kp = 0; kp < K; kp += 128) {
        // ---- cooperative stage of the 128x128 weight panel into LDS ----
        if (kp) __syncthreads();   // previous panel fully consumed
#pragma unroll
        for (int i = 0; i < 8; i++) {
            int cid = tid + 128 * i;           // 1024 chunks of 16 bf16 (32B)
            int row = cid >> 3;                // weight column 0..127
            int off = (cid & 7) * 16;          // K offset within panel
            *(v16bf*)(ldsw + row * LDSW_STRIDE + off) =
                *(const v16bf*)(wt + (long)row * K + kp + off);
        }
        __syncthreads();

        // prefetch next A panel (global_prefetch_b8)
        if (kp + 128 < K) {
            __builtin_prefetch((const void*)(arow0 + kp + 128 + hl * 8), 0, 3);
            __builtin_prefetch((const void*)(arow1 + kp + 128 + hl * 8), 0, 3);
        }

        // ---- K-steps over the staged panel ----
#pragma unroll
        for (int kb = 0; kb < 128; kb += 32) {
            v8bf a0lo = *(const v8bf*)(arow0 + kp + kb + hl * 16);
            v8bf a0hi = *(const v8bf*)(arow0 + kp + kb + 8 + hl * 16);
            v8bf a1lo = *(const v8bf*)(arow1 + kp + kb + hl * 16);
            v8bf a1hi = *(const v8bf*)(arow1 + kp + kb + 8 + hl * 16);
            v16bf af0 = __builtin_shufflevector(a0lo, a0hi,
                0, 1, 2, 3, 4, 5, 6, 7, 8, 9, 10, 11, 12, 13, 14, 15);
            v16bf af1 = __builtin_shufflevector(a1lo, a1hi,
                0, 1, 2, 3, 4, 5, 6, 7, 8, 9, 10, 11, 12, 13, 14, 15);
#pragma unroll
            for (int nt = 0; nt < 8; nt++) {
                v16bf bf = *(const v16bf*)(ldsw + (nt * 16 + l15) * LDSW_STRIDE
                                           + kb + hl * 16);
                acc0[nt] = __builtin_amdgcn_wmma_f32_16x16x32_bf16(
                    false, af0, false, bf, (short)0, acc0[nt], false, false);
                acc1[nt] = __builtin_amdgcn_wmma_f32_16x16x32_bf16(
                    false, af1, false, bf, (short)0, acc1[nt], false, false);
            }
        }
    }

    // Epilogue: C/D layout -> VGPR r: lanes 0-15 M=r, lanes 16-31 M=r+8
#pragma unroll
    for (int nt = 0; nt < 8; nt++) {
        int ncol = nt * 16 + l15;
        float bval = bs ? bs[ncol] : 0.0f;
#pragma unroll
        for (int r = 0; r < 8; r++) {
            long m0 = tile_m + r + hl * 8;
            float v0 = acc0[nt][r] + bval;
            if (do_lrelu) v0 = lrelu(v0);
            C[m0 * ldc + ncol] = (bf16)v0;

            long m1 = tile_m + 16 + r + hl * 8;
            float v1 = acc1[nt][r] + bval;
            if (do_lrelu) v1 = lrelu(v1);
            C[m1 * ldc + ncol] = (bf16)v1;
        }
    }
}

// ---------------------------------------------------------------------------
// 5) Actor attention: two softmax-attention stages (oa: J=2, goal: J=3).
//    One block = one (agent, batch) row; LDS dot-reduce over H.
//    Writes ov0 -> merge_in[:, H:2H], ov1 -> merge_in[:, 2H:3H].
// ---------------------------------------------------------------------------
__global__ __launch_bounds__(128)
void actor_attn_kernel(const bf16* __restrict__ sel,        // [2][NA*B][H]
                       const bf16* __restrict__ akeys_oa,   // [NA*2][B][H]
                       const bf16* __restrict__ avals_oa,
                       const bf16* __restrict__ akeys_g,    // [NA*3][B][H]
                       const bf16* __restrict__ avals_g,
                       bf16* __restrict__ merge_in) {       // [NA*B][3H]
    long nb = blockIdx.x;
    int n = (int)(nb / BATCH);
    long b = nb % BATCH;
    int d = threadIdx.x;

    __shared__ float red[128];
    __shared__ float lg[3];

    // ---- stage 0: other-agent attention (J = 2) ----
    {
        float sv = (float)sel[nb * H + d];
        for (int j = 0; j < 2; j++) {
            float kv = (float)akeys_oa[(((long)(n * 2 + j)) * BATCH + b) * H + d];
            red[d] = sv * kv;
            __syncthreads();
            for (int w = 64; w > 0; w >>= 1) {
                if (d < w) red[d] += red[d + w];
                __syncthreads();
            }
            if (d == 0) lg[j] = red[0] * ATTN_SCALE;
            __syncthreads();
        }
        float mx = fmaxf(lg[0], lg[1]);
        float e0 = __expf(lg[0] - mx), e1 = __expf(lg[1] - mx);
        float inv = 1.0f / (e0 + e1);
        float ov = (e0 * inv) * (float)avals_oa[(((long)(n * 2)) * BATCH + b) * H + d]
                 + (e1 * inv) * (float)avals_oa[(((long)(n * 2 + 1)) * BATCH + b) * H + d];
        merge_in[nb * (3 * H) + H + d] = (bf16)ov;
    }
    __syncthreads();

    // ---- stage 1: goal attention (J = 3) ----
    {
        float sv = (float)sel[((long)NA * BATCH + nb) * H + d];
        for (int j = 0; j < 3; j++) {
            float kv = (float)akeys_g[(((long)(n * 3 + j)) * BATCH + b) * H + d];
            red[d] = sv * kv;
            __syncthreads();
            for (int w = 64; w > 0; w >>= 1) {
                if (d < w) red[d] += red[d + w];
                __syncthreads();
            }
            if (d == 0) lg[j] = red[0] * ATTN_SCALE;
            __syncthreads();
        }
        float mx = fmaxf(fmaxf(lg[0], lg[1]), lg[2]);
        float e0 = __expf(lg[0] - mx), e1 = __expf(lg[1] - mx), e2 = __expf(lg[2] - mx);
        float inv = 1.0f / (e0 + e1 + e2);
        float ov = (e0 * inv) * (float)avals_g[(((long)(n * 3)) * BATCH + b) * H + d]
                 + (e1 * inv) * (float)avals_g[(((long)(n * 3 + 1)) * BATCH + b) * H + d]
                 + (e2 * inv) * (float)avals_g[(((long)(n * 3 + 2)) * BATCH + b) * H + d];
        merge_in[nb * (3 * H) + 2 * H + d] = (bf16)ov;
    }
}

// ---------------------------------------------------------------------------
// 6) Critic cross-agent attention (attend over the 2 agents j != n; the
//    -1e9 self-mask makes the self weight exactly ~0).
//    Writes ov -> cw1_in[:, H:2H].
// ---------------------------------------------------------------------------
__global__ __launch_bounds__(128)
void critic_attn_kernel(const bf16* __restrict__ csels,   // [NA*B][H]
                        const bf16* __restrict__ ckeys,   // [NA][B][H]
                        const bf16* __restrict__ cvals,
                        bf16* __restrict__ cw1_in) {      // [NA*B][2H]
    long nb = blockIdx.x;
    int n = (int)(nb / BATCH);
    long b = nb % BATCH;
    int d = threadIdx.x;

    int j0 = (n == 0) ? 1 : 0;
    int j1 = (n == 2) ? 1 : 2;

    __shared__ float red[128];
    __shared__ float lg[2];

    float sv = (float)csels[nb * H + d];
    {
        float kv = (float)ckeys[(((long)j0) * BATCH + b) * H + d];
        red[d] = sv * kv;
        __syncthreads();
        for (int w = 64; w > 0; w >>= 1) {
            if (d < w) red[d] += red[d + w];
            __syncthreads();
        }
        if (d == 0) lg[0] = red[0] * ATTN_SCALE;
        __syncthreads();
    }
    {
        float kv = (float)ckeys[(((long)j1) * BATCH + b) * H + d];
        red[d] = sv * kv;
        __syncthreads();
        for (int w = 64; w > 0; w >>= 1) {
            if (d < w) red[d] += red[d + w];
            __syncthreads();
        }
        if (d == 0) lg[1] = red[0] * ATTN_SCALE;
        __syncthreads();
    }
    float mx = fmaxf(lg[0], lg[1]);
    float e0 = __expf(lg[0] - mx), e1 = __expf(lg[1] - mx);
    float inv = 1.0f / (e0 + e1);
    float ov = (e0 * inv) * (float)cvals[(((long)j0) * BATCH + b) * H + d]
             + (e1 * inv) * (float)cvals[(((long)j1) * BATCH + b) * H + d];
    cw1_in[nb * (2 * H) + H + d] = (bf16)ov;
}

// ---------------------------------------------------------------------------
// 7) Output head: q[n,b,o] = h[n,b,:] . cW2[n,:,o] + cb2[n,o]   (f32 out)
// ---------------------------------------------------------------------------
__global__ __launch_bounds__(256)
void q_kernel(const bf16* __restrict__ h_buf, const float* __restrict__ cW2,
              const float* __restrict__ cb2, float* __restrict__ out) {
    long idx = (long)blockIdx.x * 256 + threadIdx.x;
    if (idx >= (long)NA * BATCH * ADIM) return;
    int o = (int)(idx & 1);
    long nb = idx >> 1;
    int n = (int)(nb / BATCH);
    float acc = cb2[n * ADIM + o];
    const bf16* hrow = h_buf + nb * H;
    const float* w = cW2 + (long)n * H * ADIM + o;
#pragma unroll 8
    for (int k = 0; k < H; k++) acc += (float)hrow[k] * w[(long)k * ADIM];
    out[idx] = acc;
}

// ---------------------------------------------------------------------------
// Host launcher
// ---------------------------------------------------------------------------
static inline void launch_gemm(int K, long rows, hipStream_t st,
                               const bf16* A, long lda, const bf16* Wt,
                               const float* bias, long rpg, long wstr, long bstr,
                               bf16* C, long ldc, int relu) {
    dim3 grid((unsigned)(rows / 128));
    if (K == 128)
        gemm_bf16_kernel<128><<<grid, 128, 0, st>>>(A, lda, Wt, bias, rpg, wstr, bstr, C, ldc, relu);
    else if (K == 256)
        gemm_bf16_kernel<256><<<grid, 128, 0, st>>>(A, lda, Wt, bias, rpg, wstr, bstr, C, ldc, relu);
    else
        gemm_bf16_kernel<384><<<grid, 128, 0, st>>>(A, lda, Wt, bias, rpg, wstr, bstr, C, ldc, relu);
}

extern "C" void kernel_launch(void* const* d_in, const int* in_sizes, int n_in,
                              void* d_out, int out_size, void* d_ws, size_t ws_size,
                              hipStream_t stream) {
    (void)in_sizes; (void)n_in; (void)out_size; (void)ws_size;
    const float* s       = (const float*)d_in[0];
    const float* a       = (const float*)d_in[1];
    const float* en_W    = (const float*)d_in[2];
    const float* en_b    = (const float*)d_in[3];
    const float* oa_W    = (const float*)d_in[4];
    const float* oa_b    = (const float*)d_in[5];
    const float* goal_W  = (const float*)d_in[6];
    const float* goal_b  = (const float*)d_in[7];
    const float* akey_W  = (const float*)d_in[8];
    const float* asel_W  = (const float*)d_in[9];
    const float* aval_W  = (const float*)d_in[10];
    const float* aval_b  = (const float*)d_in[11];
    const float* merge_W = (const float*)d_in[12];
    const float* merge_b = (const float*)d_in[13];
    const float* senc_W  = (const float*)d_in[14];
    const float* senc_b  = (const float*)d_in[15];
    const float* ckey_W  = (const float*)d_in[16];
    const float* csel_W  = (const float*)d_in[17];
    const float* cval_W  = (const float*)d_in[18];
    const float* cval_b  = (const float*)d_in[19];
    const float* cW1     = (const float*)d_in[20];
    const float* cb1     = (const float*)d_in[21];
    const float* cW2     = (const float*)d_in[22];
    const float* cb2     = (const float*)d_in[23];
    float* out = (float*)d_out;

    // ---- workspace carving (deterministic offsets) ----
    char* ws = (char*)d_ws;
    size_t off = 0;
    auto carve = [&](size_t bytes) -> void* {
        void* p = ws + off;
        off = (off + bytes + 255) & ~(size_t)255;
        return p;
    };
    const long NB = (long)NA * BATCH;

    float* mean = (float*)carve(NA * NCOLS * sizeof(float));
    float* rstd = (float*)carve(NA * NCOLS * sizeof(float));

    bf16* aselWt  = (bf16*)carve(2 * H * H * sizeof(bf16));
    bf16* akeyWt  = (bf16*)carve(2 * H * H * sizeof(bf16));
    bf16* avalWt  = (bf16*)carve(2 * H * H * sizeof(bf16));
    bf16* mergeWt = (bf16*)carve((size_t)NA * 3 * H * H * sizeof(bf16));
    bf16* ckeyWt  = (bf16*)carve(H * H * sizeof(bf16));
    bf16* cselWt  = (bf16*)carve(H * H * sizeof(bf16));
    bf16* cvalWt  = (bf16*)carve(H * H * sizeof(bf16));
    bf16* cw1Wt   = (bf16*)carve((size_t)NA * 2 * H * H * sizeof(bf16));

    bf16* merge_in = (bf16*)carve((size_t)NB * 3 * H * sizeof(bf16));  // [en|ov0|ov1]
    bf16* cw1_in   = (bf16*)carve((size_t)NB * 2 * H * sizeof(bf16));  // [s_enc|ov]
    bf16* oa_enc   = (bf16*)carve((size_t)NA * 2 * BATCH * H * sizeof(bf16));
    bf16* g_enc    = (bf16*)carve((size_t)NA * 3 * BATCH * H * sizeof(bf16));
    bf16* akeys_oa = (bf16*)carve((size_t)NA * 2 * BATCH * H * sizeof(bf16));
    bf16* avals_oa = (bf16*)carve((size_t)NA * 2 * BATCH * H * sizeof(bf16));
    bf16* akeys_g  = (bf16*)carve((size_t)NA * 3 * BATCH * H * sizeof(bf16));
    bf16* avals_g  = (bf16*)carve((size_t)NA * 3 * BATCH * H * sizeof(bf16));
    bf16* sel_buf  = (bf16*)carve((size_t)2 * NB * H * sizeof(bf16));
    bf16* sa_enc   = (bf16*)carve((size_t)NB * H * sizeof(bf16));
    bf16* ckeys    = (bf16*)carve((size_t)NB * H * sizeof(bf16));
    bf16* cvals    = (bf16*)carve((size_t)NB * H * sizeof(bf16));
    bf16* csels    = (bf16*)carve((size_t)NB * H * sizeof(bf16));
    bf16* h_buf    = (bf16*)carve((size_t)NB * H * sizeof(bf16));

    // ---- 1) BN stats ----
    bn_stats_kernel<<<NA * NCOLS, 256, 0, stream>>>(s, a, mean, rstd);

    // ---- 2) weight convert + transpose ----
    {
        dim3 g64(64, 1, 2);
        wtrans_kernel<<<g64, 256, 0, stream>>>(asel_W, aselWt, H, H, (long)H * H, (long)H * H);
        wtrans_kernel<<<g64, 256, 0, stream>>>(akey_W, akeyWt, H, H, (long)H * H, (long)H * H);
        wtrans_kernel<<<g64, 256, 0, stream>>>(aval_W, avalWt, H, H, (long)H * H, (long)H * H);
        dim3 g1(64, 1, 1);
        wtrans_kernel<<<g1, 256, 0, stream>>>(ckey_W, ckeyWt, H, H, 0, 0);
        wtrans_kernel<<<g1, 256, 0, stream>>>(csel_W, cselWt, H, H, 0, 0);
        wtrans_kernel<<<g1, 256, 0, stream>>>(cval_W, cvalWt, H, H, 0, 0);
        dim3 gm(192, 1, NA);
        wtrans_kernel<<<gm, 256, 0, stream>>>(merge_W, mergeWt, 3 * H, H, (long)3 * H * H, (long)3 * H * H);
        dim3 gc(128, 1, NA);
        wtrans_kernel<<<gc, 256, 0, stream>>>(cW1, cw1Wt, 2 * H, H, (long)2 * H * H, (long)2 * H * H);
    }

    // ---- 3) encoders ----
    encode_kernel<<<(unsigned)NB, 128, 0, stream>>>(
        s, a, mean, rstd, en_W, en_b, oa_W, oa_b, goal_W, goal_b,
        senc_W, senc_b, merge_in, oa_enc, g_enc, cw1_in);

    // ---- 4) actor-side WMMA GEMMs ----
    // sel[m] = en_enc @ asel_W[m]           (en_enc lives at merge_in col 0, lda=3H)
    launch_gemm(H, NB, stream, merge_in, 3 * H, aselWt, nullptr,
                NB, 0, 0, sel_buf, H, 0);
    launch_gemm(H, NB, stream, merge_in, 3 * H, aselWt + (long)H * H, nullptr,
                NB, 0, 0, sel_buf + NB * H, H, 0);
    // oa keys / vals
    launch_gemm(H, (long)NA * 2 * BATCH, stream, oa_enc, H, akeyWt, nullptr,
                (long)NA * 2 * BATCH, 0, 0, akeys_oa, H, 0);
    launch_gemm(H, (long)NA * 2 * BATCH, stream, oa_enc, H, avalWt, aval_b,
                (long)NA * 2 * BATCH, 0, 0, avals_oa, H, 1);
    // goal keys / vals
    launch_gemm(H, (long)NA * 3 * BATCH, stream, g_enc, H, akeyWt + (long)H * H, nullptr,
                (long)NA * 3 * BATCH, 0, 0, akeys_g, H, 0);
    launch_gemm(H, (long)NA * 3 * BATCH, stream, g_enc, H, avalWt + (long)H * H, aval_b + H,
                (long)NA * 3 * BATCH, 0, 0, avals_g, H, 1);

    // ---- 5) actor attention (fills merge_in cols H..3H) ----
    actor_attn_kernel<<<(unsigned)NB, 128, 0, stream>>>(
        sel_buf, akeys_oa, avals_oa, akeys_g, avals_g, merge_in);

    // ---- 6) merge: sa_enc = lrelu(merge_in @ merge_W[n] + merge_b[n]) ----
    launch_gemm(3 * H, NB, stream, merge_in, 3 * H, mergeWt, merge_b,
                BATCH, (long)3 * H * H, H, sa_enc, H, 1);

    // ---- 7) critic keys / vals / sels ----
    launch_gemm(H, NB, stream, sa_enc, H, ckeyWt, nullptr, NB, 0, 0, ckeys, H, 0);
    launch_gemm(H, NB, stream, sa_enc, H, cvalWt, cval_b, NB, 0, 0, cvals, H, 1);
    launch_gemm(H, NB, stream, cw1_in, 2 * H, cselWt, nullptr, NB, 0, 0, csels, H, 0);

    // ---- 8) critic attention (fills cw1_in cols H..2H) ----
    critic_attn_kernel<<<(unsigned)NB, 128, 0, stream>>>(csels, ckeys, cvals, cw1_in);

    // ---- 9) h = lrelu(cw1_in @ cW1[n] + cb1[n]) ----
    launch_gemm(2 * H, NB, stream, cw1_in, 2 * H, cw1Wt, cb1,
                BATCH, (long)2 * H * H, H, h_buf, H, 1);

    // ---- 10) q head ----
    {
        long total = (long)NA * BATCH * ADIM;
        q_kernel<<<(unsigned)((total + 255) / 256), 256, 0, stream>>>(h_buf, cW2, cb2, out);
    }
}
